// SelfAttentionEncoder_80418967650603
// MI455X (gfx1250) — compile-verified, matrix operand order checked by reading, not run on hardware
//
#include <hip/hip_runtime.h>
#include <hip/hip_bf16.h>
#include <math.h>

// ---------------------------------------------------------------------------
// CDNA5 WMMA types / fragments
// ---------------------------------------------------------------------------
typedef __attribute__((ext_vector_type(16))) _Float16 v16h;
typedef __attribute__((ext_vector_type(8)))  float    v8f;

union AFrag { v16h v; _Float16 h[16]; };
union CFrag { v8f  v; float     f[8];  };

// K index inside a 16x32 f16 A (or 32x16 B) fragment for element e of lane.
__device__ __forceinline__ int frag_k(int e, int hi /*lane>>4*/) {
  return ((e >> 3) << 4) + (hi << 3) + (e & 7);
}

static inline int cdiv(int a, int b) { return (a + b - 1) / b; }

// ---------------------------------------------------------------------------
// Pack W[K,N] (f32, row-major) into fragment-ready f16 tiles:
// Wp[((kt*Ntiles + nt)*32 + lane)*16 + e] = W[kt*32 + frag_k(e,lane>>4)][nt*16 + lane&15]
// Zero-padded on K/N edges so the GEMM needs no B-side predication at all.
// grid = (Ktiles, Ntiles), 32 threads.
// ---------------------------------------------------------------------------
__global__ __launch_bounds__(32)
void k_packW(const float* __restrict__ W, _Float16* __restrict__ Wp,
             int K, int N, int Ntiles) {
  int kt = blockIdx.x, nt = blockIdx.y;
  int lane = threadIdx.x;
  int lo = lane & 15, hi = lane >> 4;
  int n = nt * 16 + lo;
  _Float16* dst = Wp + (((size_t)kt * Ntiles + nt) * 32 + lane) * 16;
#pragma unroll
  for (int e = 0; e < 16; ++e) {
    int kk = kt * 32 + frag_k(e, hi);
    float v = (kk < K && n < N) ? W[(size_t)kk * N + n] : 0.0f;
    dst[e] = (_Float16)v;
  }
}

// ---------------------------------------------------------------------------
// Generic WMMA GEMM: C[M,N] = act(A[M,K] @ W[K,N] + bias), W pre-packed f16.
// One wave per 16x16 output tile. Inner loop: 4x b128 A loads (fast path) +
// one 32B v16h B load + v_wmma_f32_16x16x32_f16.  act: 0 = none, 1 = relu
// ---------------------------------------------------------------------------
__global__ __launch_bounds__(32)
void k_gemm(const float* __restrict__ A, const _Float16* __restrict__ Wp,
            const float* __restrict__ bias, float* __restrict__ C,
            int M, int K, int N, int act, int Ntiles) {
  int lane = threadIdx.x;
  int row0 = blockIdx.x * 16;
  int nt   = blockIdx.y;
  int col0 = nt * 16;
  int lo = lane & 15;
  int hi = lane >> 4;
  int m = row0 + lo;
  int n = col0 + lo;

  CFrag acc;
#pragma unroll
  for (int r = 0; r < 8; ++r) acc.f[r] = 0.0f;

  bool rowfull = (row0 + 16 <= M);
  int Ktiles = (K + 31) >> 5;
  const float* arow = A + (size_t)m * K;

  for (int kt = 0; kt < Ktiles; ++kt) {
    int k0 = kt << 5;
    AFrag a, b;
    // B fragment: one contiguous 32-byte load from the packed tile
    b.v = *(const v16h*)(Wp + (((size_t)kt * Ntiles + nt) * 32 + lane) * 16);

    if (rowfull && k0 + 32 <= K) {
      if (k0 + 64 <= K) {
        __builtin_prefetch(arow + k0 + 32, 0, 1);  // global_prefetch_b8
      }
      const float4* ap0 = (const float4*)(arow + k0 + hi * 8);
      const float4* ap1 = (const float4*)(arow + k0 + 16 + hi * 8);
      float4 x0 = ap0[0], x1 = ap0[1];
      float4 x2 = ap1[0], x3 = ap1[1];
      a.h[0]  = (_Float16)x0.x; a.h[1]  = (_Float16)x0.y;
      a.h[2]  = (_Float16)x0.z; a.h[3]  = (_Float16)x0.w;
      a.h[4]  = (_Float16)x1.x; a.h[5]  = (_Float16)x1.y;
      a.h[6]  = (_Float16)x1.z; a.h[7]  = (_Float16)x1.w;
      a.h[8]  = (_Float16)x2.x; a.h[9]  = (_Float16)x2.y;
      a.h[10] = (_Float16)x2.z; a.h[11] = (_Float16)x2.w;
      a.h[12] = (_Float16)x3.x; a.h[13] = (_Float16)x3.y;
      a.h[14] = (_Float16)x3.z; a.h[15] = (_Float16)x3.w;
    } else {
#pragma unroll
      for (int e = 0; e < 16; ++e) {
        int kk = k0 + frag_k(e, hi);
        a.h[e] = (_Float16)((kk < K && m < M) ? A[(size_t)m * K + kk] : 0.0f);
      }
    }
    acc.v = __builtin_amdgcn_wmma_f32_16x16x32_f16(
        false, a.v, false, b.v, (short)0, acc.v, false, false);
  }

#pragma unroll
  for (int r = 0; r < 8; ++r) {
    int mr = row0 + (hi << 3) + r;
    if (mr < M && n < N) {
      float v = acc.f[r];
      if (bias) v += bias[n];
      if (act == 1) v = fmaxf(v, 0.0f);
      C[(size_t)mr * N + n] = v;
    }
  }
}

// ---------------------------------------------------------------------------
// Flash attention, head-dim 16, seq S. qkv layout [S, 384] (Q|K|V, each 128
// wide, head h occupies cols h*16..h*16+15). out [S,128] head-concat.
// One wave per (16-query tile, head). Scores and P@V both via WMMA; online
// softmax with shfl_xor reductions over the 16-lane C-fragment row groups.
// ---------------------------------------------------------------------------
__global__ __launch_bounds__(32)
void k_attn(const float* __restrict__ qkv, float* __restrict__ out, int S) {
  const int D3 = 384;
  int lane = threadIdx.x;
  int h    = blockIdx.y;
  int q0   = blockIdx.x * 16;
  int lo   = lane & 15;
  int hi   = lane >> 4;
  const float scale = 0.25f;  // 1/sqrt(16)

  // Q fragment: per-lane contiguous 8 floats at kk = hi*8 .. hi*8+7;
  // elements 8..15 (kk >= 16) are the zero K-padding.
  AFrag aq;
  {
    const float4* qp = (const float4*)(qkv + (size_t)(q0 + lo) * D3 + h * 16 + hi * 8);
    float4 q0v = qp[0], q1v = qp[1];
    aq.h[0] = (_Float16)(q0v.x * scale); aq.h[1] = (_Float16)(q0v.y * scale);
    aq.h[2] = (_Float16)(q0v.z * scale); aq.h[3] = (_Float16)(q0v.w * scale);
    aq.h[4] = (_Float16)(q1v.x * scale); aq.h[5] = (_Float16)(q1v.y * scale);
    aq.h[6] = (_Float16)(q1v.z * scale); aq.h[7] = (_Float16)(q1v.w * scale);
#pragma unroll
    for (int e = 8; e < 16; ++e) aq.h[e] = (_Float16)0.0f;
  }

  CFrag acc;
  float rmax[8], rsum[8];
#pragma unroll
  for (int r = 0; r < 8; ++r) { acc.f[r] = 0.0f; rmax[r] = -3.0e38f; rsum[r] = 0.0f; }

  __shared__ float P[256];

  for (int kb = 0; kb < S; kb += 16) {
    // K^T fragment (B matrix: rows = dims (pad->32), cols = 16 keys)
    AFrag bk;
    {
      const float4* kp = (const float4*)(qkv + (size_t)(kb + lo) * D3 + 128 + h * 16 + hi * 8);
      float4 k0v = kp[0], k1v = kp[1];
      bk.h[0] = (_Float16)k0v.x; bk.h[1] = (_Float16)k0v.y;
      bk.h[2] = (_Float16)k0v.z; bk.h[3] = (_Float16)k0v.w;
      bk.h[4] = (_Float16)k1v.x; bk.h[5] = (_Float16)k1v.y;
      bk.h[6] = (_Float16)k1v.z; bk.h[7] = (_Float16)k1v.w;
#pragma unroll
      for (int e = 8; e < 16; ++e) bk.h[e] = (_Float16)0.0f;
    }
    CFrag s;
#pragma unroll
    for (int r = 0; r < 8; ++r) s.f[r] = 0.0f;
    s.v = __builtin_amdgcn_wmma_f32_16x16x32_f16(
        false, aq.v, false, bk.v, (short)0, s.v, false, false);

    // online softmax per row (rows live across 16-lane halves)
#pragma unroll
    for (int r = 0; r < 8; ++r) {
      float x = s.f[r];
      float tm = x;
      tm = fmaxf(tm, __shfl_xor(tm, 1, 32));
      tm = fmaxf(tm, __shfl_xor(tm, 2, 32));
      tm = fmaxf(tm, __shfl_xor(tm, 4, 32));
      tm = fmaxf(tm, __shfl_xor(tm, 8, 32));
      float nm = fmaxf(rmax[r], tm);
      float al = __expf(rmax[r] - nm);
      float p  = __expf(x - nm);
      float ps = p;
      ps += __shfl_xor(ps, 1, 32);
      ps += __shfl_xor(ps, 2, 32);
      ps += __shfl_xor(ps, 4, 32);
      ps += __shfl_xor(ps, 8, 32);
      rsum[r] = rsum[r] * al + ps;
      acc.f[r] *= al;
      rmax[r] = nm;
      P[((hi << 3) + r) * 16 + lo] = p;  // C layout -> row-major LDS
    }
    __syncthreads();

    // P as A fragment (vector LDS read), V as B fragment
    AFrag ap, bv;
    {
      const float4* pp = (const float4*)(&P[lo * 16 + hi * 8]);
      float4 p0 = pp[0], p1 = pp[1];
      ap.h[0] = (_Float16)p0.x; ap.h[1] = (_Float16)p0.y;
      ap.h[2] = (_Float16)p0.z; ap.h[3] = (_Float16)p0.w;
      ap.h[4] = (_Float16)p1.x; ap.h[5] = (_Float16)p1.y;
      ap.h[6] = (_Float16)p1.z; ap.h[7] = (_Float16)p1.w;
#pragma unroll
      for (int e = 8; e < 16; ++e) ap.h[e] = (_Float16)0.0f;
    }
#pragma unroll
    for (int e = 0; e < 8; ++e) {
      int kk = (hi << 3) + e;  // key row within tile
      bv.h[e] = (_Float16)qkv[(size_t)(kb + kk) * D3 + 256 + h * 16 + lo];
    }
#pragma unroll
    for (int e = 8; e < 16; ++e) bv.h[e] = (_Float16)0.0f;

    acc.v = __builtin_amdgcn_wmma_f32_16x16x32_f16(
        false, ap.v, false, bv.v, (short)0, acc.v, false, false);
    __syncthreads();
  }

#pragma unroll
  for (int r = 0; r < 8; ++r) {
    int row = q0 + (hi << 3) + r;
    out[(size_t)row * 128 + h * 16 + lo] = acc.f[r] / rsum[r];
  }
}

// ---------------------------------------------------------------------------
// y = LayerNorm(a + b) * g + bt ; one block of 128 threads per row (D=128)
// ---------------------------------------------------------------------------
__global__ void k_add_ln(const float* __restrict__ a, const float* __restrict__ b,
                         const float* __restrict__ g, const float* __restrict__ bt,
                         float* __restrict__ y) {
  int row = blockIdx.x, t = threadIdx.x;
  __shared__ float sh[128];
  float x = a[(size_t)row * 128 + t] + b[(size_t)row * 128 + t];
  sh[t] = x; __syncthreads();
  for (int o = 64; o > 0; o >>= 1) { if (t < o) sh[t] += sh[t + o]; __syncthreads(); }
  float mu = sh[0] * (1.0f / 128.0f); __syncthreads();
  float dx = x - mu;
  sh[t] = dx * dx; __syncthreads();
  for (int o = 64; o > 0; o >>= 1) { if (t < o) sh[t] += sh[t + o]; __syncthreads(); }
  float var = sh[0] * (1.0f / 128.0f);
  y[(size_t)row * 128 + t] = dx * rsqrtf(var + 1e-5f) * g[t] + bt[t];
}

// ---------------------------------------------------------------------------
// workflow MLP: 4 -> 128 -> 128 -> 128, one block of 128 threads
// ---------------------------------------------------------------------------
__global__ void k_wf(const float* rt, const float* w1, const float* b1,
                     const float* w2, const float* b2,
                     const float* w3, const float* b3, float* wf) {
  int t = threadIdx.x;
  __shared__ float h1[128], h2[128];
  float v = rt[0] * w1[t] + rt[1] * w1[128 + t] + rt[2] * w1[256 + t] + rt[3] * w1[384 + t] + b1[t];
  h1[t] = fmaxf(v, 0.0f); __syncthreads();
  float a = b2[t];
  for (int i = 0; i < 128; ++i) a += h1[i] * w2[i * 128 + t];
  h2[t] = fmaxf(a, 0.0f); __syncthreads();
  float c = b3[t];
  for (int i = 0; i < 128; ++i) c += h2[i] * w3[i * 128 + t];
  wf[t] = c;
}

// ---------------------------------------------------------------------------
// dag_x per-feature mean / std (ddof=1); gridDim.x = 6 features
// ---------------------------------------------------------------------------
__global__ void k_stats(const float* __restrict__ x, float* __restrict__ ms, int N) {
  int f = blockIdx.x, t = threadIdx.x;
  __shared__ float s1[256], s2[256];
  float a = 0.0f, b = 0.0f;
  for (int i = t; i < N; i += 256) { float v = x[(size_t)i * 6 + f]; a += v; b += v * v; }
  s1[t] = a; s2[t] = b; __syncthreads();
  for (int o = 128; o > 0; o >>= 1) {
    if (t < o) { s1[t] += s1[t + o]; s2[t] += s2[t + o]; }
    __syncthreads();
  }
  if (t == 0) {
    float mu  = s1[0] / N;
    float var = (s2[0] - N * mu * mu) / (N - 1);
    ms[f] = mu;
    ms[6 + f] = sqrtf(fmaxf(var, 0.0f));
  }
}

__global__ void k_standardize(const float* __restrict__ x, const float* __restrict__ ms,
                              float* __restrict__ y, int total) {
  int i = blockIdx.x * blockDim.x + threadIdx.x;
  if (i >= total) return;
  int f = i % 6;
  y[i] = (x[i] - ms[f]) / ms[6 + f];
}

// ---------------------------------------------------------------------------
// edge list materialization
// ---------------------------------------------------------------------------
__global__ void k_edges_dag(const int* __restrict__ ei, int E, int N,
                            int* __restrict__ src, int* __restrict__ dst) {
  int t = blockIdx.x * blockDim.x + threadIdx.x;
  int T = 2 * E + N;
  if (t >= T) return;
  int s, d;
  if (t < E)            { s = ei[t];           d = ei[E + t];   }
  else if (t < 2 * E)   { s = ei[E + (t - E)]; d = ei[t - E];   }
  else                  { s = d = t - 2 * E; }
  src[t] = s; dst[t] = d;
}

__global__ void k_edges_star(int M, int* __restrict__ src, int* __restrict__ dst) {
  int t = blockIdx.x * blockDim.x + threadIdx.x;
  int T = 3 * M;
  if (t >= T) return;
  int s, d;
  if (t < M)        { s = 0;       d = t;       }
  else if (t < 2*M) { s = t - M;   d = 0;       }
  else              { s = d = t - 2 * M; }
  src[t] = s; dst[t] = d;
}

// ---------------------------------------------------------------------------
// GAT helpers
// ---------------------------------------------------------------------------
__device__ __forceinline__ void atomicMaxF(float* addr, float val) {
  if (val >= 0.0f) atomicMax((int*)addr, __float_as_int(val));
  else             atomicMin((unsigned int*)addr, __float_as_uint(val));
}

__global__ void k_fill(float* __restrict__ p, int n, float v) {
  int i = blockIdx.x * blockDim.x + threadIdx.x;
  if (i < n) p[i] = v;
}

// s[n,h] = <h_feat[n,h,:], a_src[h]>, d[n,h] = <h_feat[n,h,:], a_dst[h]>
__global__ void k_sd(const float* __restrict__ hf, const float* __restrict__ as,
                     const float* __restrict__ ad, float* __restrict__ s,
                     float* __restrict__ d, int N, int H) {
  int idx = blockIdx.x * blockDim.x + threadIdx.x;
  if (idx >= N * H) return;
  int n = idx / H, hh = idx % H;
  const float* hp = hf + (size_t)n * H * 128 + hh * 128;
  const float* ap = as + hh * 128;
  const float* dp = ad + hh * 128;
  float sv = 0.0f, dv = 0.0f;
  for (int c = 0; c < 128; ++c) { float v = hp[c]; sv += v * ap[c]; dv += v * dp[c]; }
  s[idx] = sv; d[idx] = dv;
}

__device__ __forceinline__ float leaky02(float v) { return v > 0.0f ? v : 0.2f * v; }

__global__ void k_edge_max(const int* __restrict__ src, const int* __restrict__ dst,
                           const float* __restrict__ s, const float* __restrict__ d,
                           float* __restrict__ m, int E, int H) {
  int t = blockIdx.x * blockDim.x + threadIdx.x;
  if (t >= E * H) return;
  int e = t / H, hh = t % H;
  float v = leaky02(s[src[e] * H + hh] + d[dst[e] * H + hh]);
  atomicMaxF(&m[dst[e] * H + hh], v);
}

__global__ void k_edge_den(const int* __restrict__ src, const int* __restrict__ dst,
                           const float* __restrict__ s, const float* __restrict__ d,
                           const float* __restrict__ m, float* __restrict__ den,
                           int E, int H) {
  int t = blockIdx.x * blockDim.x + threadIdx.x;
  if (t >= E * H) return;
  int e = t / H, hh = t % H;
  float v = leaky02(s[src[e] * H + hh] + d[dst[e] * H + hh]);
  atomicAdd(&den[dst[e] * H + hh], __expf(v - m[dst[e] * H + hh]));
}

// out[dst,h,:] += alpha * h_feat[src,h,:], 32-channel chunk per thread
__global__ void k_edge_agg(const int* __restrict__ src, const int* __restrict__ dst,
                           const float* __restrict__ s, const float* __restrict__ d,
                           const float* __restrict__ m, const float* __restrict__ den,
                           const float* __restrict__ hf, float* __restrict__ out,
                           int E, int H) {
  int t = blockIdx.x * blockDim.x + threadIdx.x;
  int total = E * H * 4;
  if (t >= total) return;
  int ch = t & 3;
  int hh = (t >> 2) % H;
  int e  = t / (4 * H);
  int si = src[e], di = dst[e];
  float v  = leaky02(s[si * H + hh] + d[di * H + hh]);
  float al = __expf(v - m[di * H + hh]) / den[di * H + hh];
  const float* hp = hf  + (size_t)si * H * 128 + hh * 128 + ch * 32;
  float*       op = out + (size_t)di * H * 128 + hh * 128 + ch * 32;
  for (int c = 0; c < 32; ++c) atomicAdd(&op[c], al * hp[c]);
}

// y = elu(x + bias[j % W])
__global__ void k_bias_elu(const float* __restrict__ x, const float* __restrict__ b,
                           float* __restrict__ y, int total, int W) {
  int i = blockIdx.x * blockDim.x + threadIdx.x;
  if (i >= total) return;
  float v = x[i] + b[i % W];
  y[i] = v > 0.0f ? v : expm1f(v);
}

// column mean over N rows, C columns; gridDim.x = C
__global__ void k_colmean(const float* __restrict__ x, float* __restrict__ out,
                          int N, int C) {
  int c = blockIdx.x, t = threadIdx.x;
  __shared__ float sh[256];
  float a = 0.0f;
  for (int i = t; i < N; i += 256) a += x[(size_t)i * C + c];
  sh[t] = a; __syncthreads();
  for (int o = 128; o > 0; o >>= 1) { if (t < o) sh[t] += sh[t + o]; __syncthreads(); }
  if (t == 0) out[c] = sh[0] / N;
}

// tvx[0,:] = rt, tvx[1+v,:] = vm_info[v,:]
__global__ void k_build_tvx(const float* __restrict__ rt, const float* __restrict__ vmi,
                            float* __restrict__ tvx, int M) {
  int i = blockIdx.x * blockDim.x + threadIdx.x;
  if (i >= M * 4) return;
  int r = i / 4, c = i % 4;
  tvx[i] = (r == 0) ? rt[c] : vmi[(size_t)(r - 1) * 4 + c];
}

// feat[v, 0:768] = [vm_global | wf | dag_mean | dag_ready | tv[0] | tv[1+v]]
__global__ void k_feat(const float* __restrict__ vmg, const float* __restrict__ wf,
                       const float* __restrict__ dagmean, const float* __restrict__ xd2,
                       const int* __restrict__ node_id, const float* __restrict__ tv,
                       float* __restrict__ feat, int V) {
  int i = blockIdx.x * blockDim.x + threadIdx.x;
  if (i >= V * 768) return;
  int v = i / 768, j = i % 768;
  float r;
  if (j < 128)       r = vmg[(size_t)v * 128 + j];
  else if (j < 256)  r = wf[j - 128];
  else if (j < 384)  r = dagmean[j - 256];
  else if (j < 512)  r = xd2[(size_t)(*node_id) * 128 + (j - 384)];
  else if (j < 640)  r = tv[j - 512];
  else               r = tv[(size_t)(v + 1) * 128 + (j - 640)];
  feat[i] = r;
}

// ---------------------------------------------------------------------------
// host orchestration
// ---------------------------------------------------------------------------
extern "C" void kernel_launch(void* const* d_in, const int* in_sizes, int n_in,
                              void* d_out, int out_size, void* d_ws, size_t ws_size,
                              hipStream_t stream) {
  (void)in_sizes; (void)n_in; (void)out_size; (void)ws_size;

  const int NN = 10000, NE = 100000, V = 2048, M = 2049, S = 2048;

  // inputs (setup_inputs dict order; params flattened in insertion order)
  const float* rt       = (const float*)d_in[0];
  const float* vm_info  = (const float*)d_in[1];
  const float* dag_x    = (const float*)d_in[2];
  const int*   dag_ei   = (const int*)  d_in[3];
  const int*   node_id  = (const int*)  d_in[4];
  const float* te_w1 = (const float*)d_in[5],  *te_b1 = (const float*)d_in[6];
  const float* te_w2 = (const float*)d_in[7],  *te_b2 = (const float*)d_in[8];
  const float* te_w3 = (const float*)d_in[9],  *te_b3 = (const float*)d_in[10];
  const float* vm_w  = (const float*)d_in[11], *vm_b  = (const float*)d_in[12];
  const float* eqkv_w = (const float*)d_in[13], *eqkv_b = (const float*)d_in[14];
  const float* eout_w = (const float*)d_in[15], *eout_b = (const float*)d_in[16];
  const float* ln1_g  = (const float*)d_in[17], *ln1_b  = (const float*)d_in[18];
  const float* ln2_g  = (const float*)d_in[19], *ln2_b  = (const float*)d_in[20];
  const float* f1_w   = (const float*)d_in[21], *f1_b   = (const float*)d_in[22];
  const float* f2_w   = (const float*)d_in[23], *f2_b   = (const float*)d_in[24];
  const float* gd1_w  = (const float*)d_in[25], *gd1_as = (const float*)d_in[26];
  const float* gd1_ad = (const float*)d_in[27], *gd1_b  = (const float*)d_in[28];
  const float* gd2_w  = (const float*)d_in[29], *gd2_as = (const float*)d_in[30];
  const float* gd2_ad = (const float*)d_in[31], *gd2_b  = (const float*)d_in[32];
  const float* gv1_w  = (const float*)d_in[33], *gv1_as = (const float*)d_in[34];
  const float* gv1_ad = (const float*)d_in[35], *gv1_b  = (const float*)d_in[36];
  const float* gv2_w  = (const float*)d_in[37], *gv2_as = (const float*)d_in[38];
  const float* gv2_ad = (const float*)d_in[39], *gv2_b  = (const float*)d_in[40];
  const float* pr_w1  = (const float*)d_in[41], *pr_b1  = (const float*)d_in[42];
  const float* pr_w2  = (const float*)d_in[43], *pr_b2  = (const float*)d_in[44];
  const float* pr_w3  = (const float*)d_in[45], *pr_b3  = (const float*)d_in[46];
  float* out = (float*)d_out;

  // bump allocator over workspace; every region rounded to 32B so packed
  // v16h / float4 loads stay alignment-safe.
  float* w = (float*)d_ws;
  auto alloc = [&](size_t n) { n = (n + 7) & ~(size_t)7; float* p = w; w += n; return p; };

  float* wf      = alloc(128);
  float* stats   = alloc(16);
  float* dagmean = alloc(128);
  _Float16* packW = (_Float16*)alloc(49152);  // 98304 halves, max packed weight
  float* xenc    = alloc((size_t)V * 128);   // encoder x / vm_global
  float* qkv     = alloc((size_t)V * 384);
  float* attn_o  = alloc((size_t)V * 128);   // later: pm2
  float* proj    = alloc((size_t)V * 128);
  float* ffh     = alloc((size_t)V * 512);   // later: pm1
  float* xd      = alloc((size_t)NN * 6);
  float* hbig    = alloc((size_t)NN * 512);  // dag h1; later h2 + agg2/xd2
  float* aggbig  = alloc((size_t)NN * 512);  // dag agg1/xg1; later feat
  float* sN      = alloc((size_t)NN * 4);
  float* dN      = alloc((size_t)NN * 4);
  float* mN      = alloc((size_t)NN * 4);
  float* denN    = alloc((size_t)NN * 4);
  float* tvx     = alloc((size_t)M * 4);
  float* thbig   = alloc((size_t)M * 512);   // star h1; later h2 + tv
  float* taggbig = alloc((size_t)M * 512);   // star agg1/xg1
  float* sM      = alloc((size_t)M * 4);
  float* dM      = alloc((size_t)M * 4);
  float* mM      = alloc((size_t)M * 4);
  float* denM    = alloc((size_t)M * 4);
  int* dag_src   = (int*)alloc(2 * NE + NN);
  int* dag_dst   = (int*)alloc(2 * NE + NN);
  int* st_src    = (int*)alloc(3 * M);
  int* st_dst    = (int*)alloc(3 * M);

  const int Edag = 2 * NE + NN;
  const int Est  = 3 * M;

  auto gemm = [&](const float* A, const float* Wt, const float* bias, float* C,
                  int Mm, int Kk, int Nn, int act) {
    int Ktiles = cdiv(Kk, 32), Ntiles = cdiv(Nn, 16);
    dim3 gp(Ktiles, Ntiles);
    k_packW<<<gp, 32, 0, stream>>>(Wt, packW, Kk, Nn, Ntiles);
    dim3 g(cdiv(Mm, 16), Ntiles);
    k_gemm<<<g, 32, 0, stream>>>(A, packW, bias, C, Mm, Kk, Nn, act, Ntiles);
  };

  auto run_gat = [&](const float* xin, int N, int Kin, int H,
                     const float* Wg, const float* as, const float* ad, const float* bg,
                     const int* esrc, const int* edst, int Etot,
                     float* hbuf, float* aggbuf,
                     float* sb, float* db, float* mb, float* denb, float* xout) {
    int HC = H * 128;
    gemm(xin, Wg, nullptr, hbuf, N, Kin, HC, 0);
    int nh = N * H;
    k_sd<<<cdiv(nh, 256), 256, 0, stream>>>(hbuf, as, ad, sb, db, N, H);
    k_fill<<<cdiv(nh, 256), 256, 0, stream>>>(mb, nh, -3.0e38f);
    k_fill<<<cdiv(nh, 256), 256, 0, stream>>>(denb, nh, 0.0f);
    k_fill<<<cdiv(N * HC, 256), 256, 0, stream>>>(aggbuf, N * HC, 0.0f);
    k_edge_max<<<cdiv(Etot * H, 256), 256, 0, stream>>>(esrc, edst, sb, db, mb, Etot, H);
    k_edge_den<<<cdiv(Etot * H, 256), 256, 0, stream>>>(esrc, edst, sb, db, mb, denb, Etot, H);
    k_edge_agg<<<cdiv(Etot * H * 4, 256), 256, 0, stream>>>(esrc, edst, sb, db, mb, denb,
                                                            hbuf, aggbuf, Etot, H);
    k_bias_elu<<<cdiv(N * HC, 256), 256, 0, stream>>>(aggbuf, bg, xout, N * HC, HC);
  };

  // --- 0) preprocessing ---------------------------------------------------
  k_stats<<<6, 256, 0, stream>>>(dag_x, stats, NN);
  k_standardize<<<cdiv(NN * 6, 256), 256, 0, stream>>>(dag_x, stats, xd, NN * 6);
  k_edges_dag<<<cdiv(Edag, 256), 256, 0, stream>>>(dag_ei, NE, NN, dag_src, dag_dst);
  k_edges_star<<<cdiv(Est, 256), 256, 0, stream>>>(M, st_src, st_dst);
  k_wf<<<1, 128, 0, stream>>>(rt, te_w1, te_b1, te_w2, te_b2, te_w3, te_b3, wf);

  // --- 1) VM embedding + transformer encoder ------------------------------
  gemm(vm_info, vm_w, vm_b, xenc, V, 4, 128, 0);
  for (int l = 0; l < 2; ++l) {
    gemm(xenc, eqkv_w + (size_t)l * 128 * 384, eqkv_b + l * 384, qkv, S, 128, 384, 0);
    dim3 ga(S / 16, 8);
    k_attn<<<ga, 32, 0, stream>>>(qkv, attn_o, S);
    gemm(attn_o, eout_w + (size_t)l * 128 * 128, eout_b + l * 128, proj, S, 128, 128, 0);
    k_add_ln<<<S, 128, 0, stream>>>(xenc, proj, ln1_g + l * 128, ln1_b + l * 128, xenc);
    gemm(xenc, f1_w + (size_t)l * 128 * 512, f1_b + l * 512, ffh, S, 128, 512, 1);
    gemm(ffh, f2_w + (size_t)l * 512 * 128, f2_b + l * 128, proj, S, 512, 128, 0);
    k_add_ln<<<S, 128, 0, stream>>>(xenc, proj, ln2_g + l * 128, ln2_b + l * 128, xenc);
  }
  // xenc == vm_global [2048,128]

  // --- 2) GAT over DAG -----------------------------------------------------
  run_gat(xd, NN, 6, 4, gd1_w, gd1_as, gd1_ad, gd1_b,
          dag_src, dag_dst, Edag, hbig, aggbig, sN, dN, mN, denN, aggbig);
  float* h2   = hbig;
  float* agg2 = hbig + (size_t)NN * 128;
  run_gat(aggbig, NN, 512, 1, gd2_w, gd2_as, gd2_ad, gd2_b,
          dag_src, dag_dst, Edag, h2, agg2, sN, dN, mN, denN, agg2);
  float* xd2 = agg2;  // [NN,128]
  k_colmean<<<128, 256, 0, stream>>>(xd2, dagmean, NN, 128);

  // --- 3) GAT over task-VM star graph -------------------------------------
  k_build_tvx<<<cdiv(M * 4, 256), 256, 0, stream>>>(rt, vm_info, tvx, M);
  run_gat(tvx, M, 4, 4, gv1_w, gv1_as, gv1_ad, gv1_b,
          st_src, st_dst, Est, thbig, taggbig, sM, dM, mM, denM, taggbig);
  float* th2   = thbig;
  float* tagg2 = thbig + (size_t)M * 128;
  run_gat(taggbig, M, 512, 1, gv2_w, gv2_as, gv2_ad, gv2_b,
          st_src, st_dst, Est, th2, tagg2, sM, dM, mM, denM, tagg2);
  float* tv = tagg2;  // [M,128]

  // --- 4) combine + priority MLP ------------------------------------------
  float* feat = aggbig;  // [V,768], dag agg buffer is free now
  k_feat<<<cdiv(V * 768, 256), 256, 0, stream>>>(xenc, wf, dagmean, xd2, node_id, tv, feat, V);
  gemm(feat, pr_w1, pr_b1, ffh, V, 768, 128, 1);     // pm1
  gemm(ffh, pr_w2, pr_b2, attn_o, V, 128, 128, 1);   // pm2
  gemm(attn_o, pr_w3, pr_b3, out, V, 128, 1, 0);     // [V,1]
}
